// UpXiPredictorAsoftPos_9182640078992
// MI455X (gfx1250) — matrix operand order repaired and checked by reading, not compiled, not run on hardware
//
#include <hip/hip_runtime.h>
#include <hip/hip_bf16.h>
#include <math.h>

#define BB 4
#define NN 8192
#define KK 512
#define CC 256
#define NBLK 128      // N / 64
#define ROWS1 64
#define EPSF 1e-8f
#define AS 264        // padded LDS stride (halves) for activations
#define WSTR 264      // padded LDS stride (halves) for transposed weights

typedef _Float16 f16;
typedef __attribute__((ext_vector_type(16))) _Float16 v16h;
typedef __attribute__((ext_vector_type(8)))  _Float16 v8h;
typedef __attribute__((ext_vector_type(8)))  float    v8f;

// ---- workspace layout (byte offsets) ----
#define WS_PIDX   0            // int  [B*N]
#define WS_SCAL   131072       // f32  [B*N]   mN^2 / clip(rowsum)
#define WS_BOFF   262144       // f32  [B*NBLK*K] partial -> exclusive offsets
#define WS_ITOT   1310720      // f32  [B*K]   1/clip(column total)
#define WS_PPART  1318912      // f32  [B*4*N] per-strip pos01 partials
#define WS_H      1843200      // f16  [B*N*C] LN(s_pi)
#define WS_WT     18620416     // f16  [4][256][256] transposed weights

// ---- output layout (float element offsets) ----
#define OUT_XHAT  0
#define OUT_XI    98304
#define OUT_POS   196608
#define OUT_WUSE  229376

static __device__ __forceinline__ float wave_sum(float v) {
  #pragma unroll
  for (int m = 16; m >= 1; m >>= 1) v += __shfl_xor(v, m, 32);
  return v;
}

static __device__ __forceinline__ float fast_silu(float v) {
  return v * __builtin_amdgcn_rcpf(1.f + __expf(-v));
}

// ------------------------------------------------------------------
// k0: one-shot f32->f16 transpose of the four 256x256 weight matrices
// grid: 4*256 blocks of 256 threads
// ------------------------------------------------------------------
__global__ void __launch_bounds__(256) k0_wt(
    const float* __restrict__ Wq1, const float* __restrict__ Wq2,
    const float* __restrict__ Wm1, const float* __restrict__ Wm2,
    f16* __restrict__ wt) {
  int m = blockIdx.x >> 8;
  int k = blockIdx.x & 255;
  int t = threadIdx.x;
  const float* W = (m == 0) ? Wq1 : (m == 1) ? Wq2 : (m == 2) ? Wm1 : Wm2;
  // wt[m][n][k] = W[k][n]  (n-major, k contiguous)
  wt[((size_t)m << 16) + (size_t)t * 256 + k] = (f16)W[k * 256 + t];
}

// ------------------------------------------------------------------
// k1: normalize rows of A_soft, argmax -> one-hot w_use, block column sums
// grid: B*NBLK blocks of 512 threads (16 waves, 4 rows/wave)
// ------------------------------------------------------------------
__global__ void __launch_bounds__(512) k1_norm(
    const float* __restrict__ A, const float* __restrict__ mpar,
    const float* __restrict__ nmask, float* __restrict__ out,
    int* __restrict__ pidx, float* __restrict__ scal,
    float* __restrict__ bpart) {
  __shared__ float mp[KK];
  __shared__ float colS[16 * KK];
  int tid = threadIdx.x;
  int blk = blockIdx.x % NBLK;
  int b   = blockIdx.x / NBLK;
  mp[tid] = mpar[b * KK + tid];
  __syncthreads();
  int w = tid >> 5, l = tid & 31;
  int k0 = 16 * l;
  float acc[16];
  #pragma unroll
  for (int i = 0; i < 16; i++) acc[i] = 0.f;

  for (int r = 0; r < 4; r++) {
    int n = blk * ROWS1 + w * 4 + r;
    float mN = nmask[b * NN + n];
    const float4* ap = (const float4*)(A + ((size_t)(b * NN + n)) * KK + k0);
    float v[16];
    #pragma unroll
    for (int g = 0; g < 4; g++) {
      float4 q = ap[g];
      v[4*g+0] = q.x; v[4*g+1] = q.y; v[4*g+2] = q.z; v[4*g+3] = q.w;
    }
    float lsum = 0.f, bmax = -INFINITY; int bi = k0;
    #pragma unroll
    for (int i = 0; i < 16; i++) {
      float t = v[i] * mp[k0 + i] * mN;
      v[i] = t; lsum += t;
      if (t > bmax) { bmax = t; bi = k0 + i; }
    }
    float rsum = wave_sum(lsum);
    #pragma unroll
    for (int m = 16; m >= 1; m >>= 1) {          // argmax, first-index tiebreak
      float ov = __shfl_xor(bmax, m, 32);
      int   oi = __shfl_xor(bi,   m, 32);
      if (ov > bmax || (ov == bmax && oi < bi)) { bmax = ov; bi = oi; }
    }
    float inv = mN / fmaxf(rsum, EPSF);
    #pragma unroll
    for (int i = 0; i < 16; i++) acc[i] += v[i] * inv;   // column sums of w
    // one-hot w_use row
    float4* op = (float4*)(out + (size_t)OUT_WUSE + ((size_t)(b * NN + n)) * KK + k0);
    #pragma unroll
    for (int g = 0; g < 4; g++) {
      float4 z; z.x = z.y = z.z = z.w = 0.f;
      int base = k0 + 4 * g;
      if (bi >= base && bi < base + 4) {
        if      (bi == base)     z.x = mN;
        else if (bi == base + 1) z.y = mN;
        else if (bi == base + 2) z.z = mN;
        else                     z.w = mN;
      }
      op[g] = z;
    }
    if (l == 0) {
      pidx[b * NN + n] = bi;
      scal[b * NN + n] = mN * mN / fmaxf(rsum, EPSF);
    }
  }
  #pragma unroll
  for (int i = 0; i < 16; i++) colS[w * KK + k0 + i] = acc[i];
  __syncthreads();
  float s = 0.f;
  #pragma unroll
  for (int wv = 0; wv < 16; wv++) s += colS[wv * KK + tid];
  bpart[((size_t)(b * NBLK + blk)) * KK + tid] = s;
}

// ------------------------------------------------------------------
// k2: exclusive scan over blocks per (b,k); grid B x 512 threads
// ------------------------------------------------------------------
__global__ void __launch_bounds__(512) k2_scan(float* __restrict__ boff,
                                               float* __restrict__ itot) {
  int k = threadIdx.x, b = blockIdx.x;
  float off = 0.f;
  for (int blk = 0; blk < NBLK; blk++) {
    size_t idx = ((size_t)(b * NBLK + blk)) * KK + k;
    float p = boff[idx];
    boff[idx] = off;
    off += p;
  }
  itot[b * KK + k] = 1.f / fmaxf(off, EPSF);
}

// ------------------------------------------------------------------
// k3: recompute w, intra-block cum, per-strip pos01 partials
// grid: B*NBLK*4 blocks of 128 threads (one column each)
// ------------------------------------------------------------------
__global__ void __launch_bounds__(128) k3_pos(
    const float* __restrict__ A, const float* __restrict__ mpar,
    const float* __restrict__ scal, const float* __restrict__ boff,
    const float* __restrict__ itot, float* __restrict__ ppart) {
  __shared__ float contrib[ROWS1 * 129];
  int t = threadIdx.x;
  int x = blockIdx.x;
  int strip = x & 3;
  int y = x >> 2;
  int blk = y % NBLK, b = y / NBLK;
  int k = strip * 128 + t;
  float mpk = mpar[b * KK + k];
  float cum = boff[((size_t)(b * NBLK + blk)) * KK + k];
  float it  = itot[b * KK + k];
  int n0 = blk * ROWS1;
  for (int r = 0; r < ROWS1; r++) {
    int n = n0 + r;
    float a  = A[((size_t)(b * NN + n)) * KK + k];
    float wv = a * mpk * scal[b * NN + n];
    cum += wv;
    contrib[r * 129 + t] = wv * cum * it;
  }
  __syncthreads();
  if (t < ROWS1) {
    float s = 0.f;
    for (int j = 0; j < 128; j++) s += contrib[t * 129 + j];
    ppart[((size_t)(b * 4 + strip)) * NN + n0 + t] = s;
  }
}

// ------------------------------------------------------------------
// k4: pos01 out + Fourier pos-emb + s_parent gather + LN -> f16 h
// grid: (B*N)/128 blocks of 256 threads (8 waves x 16 rows)
// ------------------------------------------------------------------
__global__ void __launch_bounds__(256) k4_emb(
    const float* __restrict__ sparent, const float* __restrict__ nmask,
    const float* __restrict__ Wpos, const float* __restrict__ bpos,
    const float* __restrict__ lng, const float* __restrict__ lnb,
    const int* __restrict__ pidx, const float* __restrict__ ppart,
    f16* __restrict__ hws, float* __restrict__ out) {
  __shared__ float wpos_s[18 * CC];
  __shared__ float bpos_s[CC], g_s[CC], b_s[CC];
  int tid = threadIdx.x;
  for (int i = tid; i < 18 * CC; i += 256) wpos_s[i] = Wpos[i];
  bpos_s[tid] = bpos[tid]; g_s[tid] = lng[tid]; b_s[tid] = lnb[tid];
  __syncthreads();
  int w = tid >> 5, l = tid & 31;
  int rowBase = blockIdx.x * 128;
  for (int i = 0; i < 16; i++) {
    int grow = rowBase + w * 16 + i;
    int b = grow / NN, n = grow % NN;
    float mN = nmask[grow];
    float p = 0.f;
    #pragma unroll
    for (int s = 0; s < 4; s++) p += ppart[((size_t)(b * 4 + s)) * NN + n];
    p *= mN;
    if (l == 0) out[OUT_POS + grow] = p;
    // sin(2*pi*p*2^f): hardware v_sin/v_cos take revolutions directly
    float sn[9], cs[9];
    #pragma unroll
    for (int f = 0; f < 9; f++) {
      float rev = p * (float)(1 << f);
      sn[f] = __builtin_amdgcn_sinf(rev);
      cs[f] = __builtin_amdgcn_cosf(rev);
    }
    int idx = pidx[grow];
    const float* sp = sparent + ((size_t)(b * KK + idx)) * CC;
    const float4* spv = (const float4*)(sp + l * 8);
    float4 sp0 = spv[0], sp1 = spv[1];
    float spc[8] = {sp0.x, sp0.y, sp0.z, sp0.w, sp1.x, sp1.y, sp1.z, sp1.w};
    float xv[8]; float ls = 0.f, lq = 0.f;
    #pragma unroll
    for (int j = 0; j < 8; j++) {
      int c = l * 8 + j;
      float pe = bpos_s[c];
      #pragma unroll
      for (int f = 0; f < 9; f++)
        pe += sn[f] * wpos_s[f * CC + c] + cs[f] * wpos_s[(9 + f) * CC + c];
      pe *= mN;
      float v = (mN * spc[j] + pe) * mN;
      xv[j] = v; ls += v; lq += v * v;
    }
    float sum = wave_sum(ls), sq = wave_sum(lq);
    float mu = sum * (1.f / CC);
    float var = sq * (1.f / CC) - mu * mu;
    float rs = rsqrtf(var + 1e-5f);
    v8h hv;
    #pragma unroll
    for (int j = 0; j < 8; j++) {
      int c = l * 8 + j;
      hv[j] = (f16)((xv[j] - mu) * rs * g_s[c] + b_s[c]);
    }
    *(v8h*)(hws + ((size_t)grow) * CC + l * 8) = hv;
  }
}

// ------------------------------------------------------------------
// k5: MLP chain with WMMA f16 (f32 accum)
// ------------------------------------------------------------------
static __device__ __forceinline__ v16h combine16(v8h lo, v8h hi) {
  return __builtin_shufflevector(lo, hi, 0,1,2,3,4,5,6,7,8,9,10,11,12,13,14,15);
}

__device__ void gemm_stage(const f16* __restrict__ src, f16* __restrict__ dst,
                           f16* __restrict__ Wt, const f16* __restrict__ WtG,
                           const float* __restrict__ bg, int doSilu, int tid) {
  // stage pre-transposed f16 weights (n-major) into padded LDS, 16B vectors
  {
    const uint4* gw = (const uint4*)WtG;          // 8192 uint4
    #pragma unroll
    for (int i = 0; i < 32; i++) {
      int idx = tid + 256 * i;
      int row = idx >> 5, q = idx & 31;           // row = n, q*8 = k chunk
      *(uint4*)(Wt + row * WSTR + q * 8) = gw[idx];
    }
  }
  __syncthreads();
  int w = tid >> 5, l = tid & 31, lr = l & 15, hs = l >> 4;
  int rowb = w * 16;
  const f16* arow = src + (rowb + lr) * AS;
  v16h a[8];
  #pragma unroll
  for (int kk = 0; kk < 8; kk++) {                // A 16x32 lane layout
    int kb = kk * 32 + hs * 8;
    v8h lo = *(const v8h*)(arow + kb);
    v8h hi = *(const v8h*)(arow + kb + 16);
    a[kk] = combine16(lo, hi);
  }
  for (int nt = 0; nt < 16; nt++) {
    float bias = bg[nt * 16 + lr];
    v8f acc = {0.f, 0.f, 0.f, 0.f, 0.f, 0.f, 0.f, 0.f};
    const f16* bcol = Wt + (nt * 16 + lr) * WSTR + hs * 8;    // B 32x16 lane layout
    #pragma unroll
    for (int kk = 0; kk < 8; kk++) {
      v8h b0 = *(const v8h*)(bcol + kk * 32);
      v8h b1 = *(const v8h*)(bcol + kk * 32 + 16);
      v16h bf = combine16(b0, b1);
      acc = __builtin_amdgcn_wmma_f32_16x16x32_f16(false, a[kk], false, bf,
                                                   (short)0, acc, false, false);
    }
    #pragma unroll
    for (int j = 0; j < 8; j++) {                 // D: VGPR j -> M = j + hs*8
      float v = acc[j] + bias;
      if (doSilu) v = fast_silu(v);
      dst[(rowb + hs * 8 + j) * AS + nt * 16 + lr] = (f16)v;
    }
  }
  __syncthreads();
}

__global__ void __launch_bounds__(256) k5_mlp(
    const f16* __restrict__ hws, const f16* __restrict__ wt,
    const float* __restrict__ bq1, const float* __restrict__ bq2,
    const float* __restrict__ lng, const float* __restrict__ lnb,
    const float* __restrict__ bm1, const float* __restrict__ bm2,
    const float* __restrict__ Wm3, const float* __restrict__ bm3,
    const float* __restrict__ nmask, const int* __restrict__ pidx,
    const float* __restrict__ muk, const float* __restrict__ Rk,
    const float* __restrict__ sk, float* __restrict__ out) {
  extern __shared__ f16 dyn[];
  f16* Wt   = dyn;                  // 256*WSTR halves
  f16* actA = Wt + 256 * WSTR;      // 128*AS halves
  f16* actB = actA + 128 * AS;      // 128*AS halves
  __shared__ float wm3s[CC * 3];
  __shared__ float bm3s3[4];
  __shared__ float lng_s[CC], lnb_s[CC];
  int tid = threadIdx.x;
  for (int i = tid; i < CC * 3; i += 256) wm3s[i] = Wm3[i];
  if (tid < 3) bm3s3[tid] = bm3[tid];
  lng_s[tid] = lng[tid]; lnb_s[tid] = lnb[tid];
  int rowBase = blockIdx.x * 128;
  { // load h tile -> actA (padded)
    int r = tid >> 1, off = (tid & 1) * 128;
    const uint4* gp = (const uint4*)(hws + ((size_t)(rowBase + r)) * CC + off);
    uint4* lp = (uint4*)(actA + r * AS + off);
    #pragma unroll
    for (int i = 0; i < 16; i++) lp[i] = gp[i];
  }
  __syncthreads();
  gemm_stage(actA, actB, Wt, wt + 0 * 65536, bq1, 1, tid);
  gemm_stage(actB, actA, Wt, wt + 1 * 65536, bq2, 0, tid);
  { // q = out * mN ; LayerNorm(ln_m) ; back to actA  (rows are wave-private)
    int w = tid >> 5, l = tid & 31;
    for (int i = 0; i < 16; i++) {
      int row = w * 16 + i, grow = rowBase + row;
      float mN = nmask[grow];
      v8h hv = *(const v8h*)(actA + row * AS + l * 8);
      float x[8]; float ls = 0.f, lq = 0.f;
      #pragma unroll
      for (int j = 0; j < 8; j++) {
        float v = (float)hv[j] * mN;
        x[j] = v; ls += v; lq += v * v;
      }
      float sum = wave_sum(ls), sq = wave_sum(lq);
      float mu = sum * (1.f / CC), var = sq * (1.f / CC) - mu * mu;
      float rs = rsqrtf(var + 1e-5f);
      v8h ov;
      #pragma unroll
      for (int j = 0; j < 8; j++) {
        int c = l * 8 + j;
        ov[j] = (f16)((x[j] - mu) * rs * lng_s[c] + lnb_s[c]);
      }
      *(v8h*)(actA + row * AS + l * 8) = ov;
    }
    __syncthreads();
  }
  gemm_stage(actA, actB, Wt, wt + 2 * 65536, bm1, 1, tid);
  gemm_stage(actB, actA, Wt, wt + 3 * 65536, bm2, 1, tid);
  { // xi = h@Wm3+bm3 ; tanh ; gathered geometry
    int w = tid >> 5, l = tid & 31;
    for (int i = 0; i < 16; i++) {
      int row = w * 16 + i, grow = rowBase + row;
      v8h hv = *(const v8h*)(actA + row * AS + l * 8);
      float a0 = 0.f, a1 = 0.f, a2 = 0.f;
      #pragma unroll
      for (int j = 0; j < 8; j++) {
        int c = l * 8 + j;
        float h = (float)hv[j];
        a0 += h * wm3s[c * 3 + 0];
        a1 += h * wm3s[c * 3 + 1];
        a2 += h * wm3s[c * 3 + 2];
      }
      a0 = wave_sum(a0); a1 = wave_sum(a1); a2 = wave_sum(a2);
      if (l == 0) {
        float mN = nmask[grow];
        int b = grow / NN;
        int idx = pidx[grow];
        float xh0 = tanhf((a0 + bm3s3[0]) * mN) * 1.5f;
        float xh1 = tanhf((a1 + bm3s3[1]) * mN) * 1.5f;
        float xh2 = tanhf((a2 + bm3s3[2]) * mN) * 1.5f;
        out[OUT_XI + (size_t)grow * 3 + 0] = xh0;
        out[OUT_XI + (size_t)grow * 3 + 1] = xh1;
        out[OUT_XI + (size_t)grow * 3 + 2] = xh2;
        size_t kb = (size_t)b * KK + idx;
        const float* mu = muk + kb * 3;
        const float* ss = sk  + kb * 3;
        const float* R  = Rk  + kb * 9;
        float s0 = fmaxf(mN * ss[0], 1e-6f);
        float s1 = fmaxf(mN * ss[1], 1e-6f);
        float s2 = fmaxf(mN * ss[2], 1e-6f);
        float l0 = xh0 * s0, l1 = xh1 * s1, l2 = xh2 * s2;
        out[(size_t)grow * 3 + 0] = (mN * mu[0] + R[0]*l0 + R[1]*l1 + R[2]*l2) * mN * 10.f;
        out[(size_t)grow * 3 + 1] = (mN * mu[1] + R[3]*l0 + R[4]*l1 + R[5]*l2) * mN * 10.f;
        out[(size_t)grow * 3 + 2] = (mN * mu[2] + R[6]*l0 + R[7]*l1 + R[8]*l2) * mN * 10.f;
      }
    }
  }
}

extern "C" void kernel_launch(void* const* d_in, const int* in_sizes, int n_in,
                              void* d_out, int out_size, void* d_ws, size_t ws_size,
                              hipStream_t stream) {
  const float* s_parent = (const float*)d_in[0];
  const float* A        = (const float*)d_in[1];
  const float* mpar     = (const float*)d_in[2];
  const float* nmask    = (const float*)d_in[3];
  const float* muk      = (const float*)d_in[4];
  const float* Rk       = (const float*)d_in[5];
  const float* sk       = (const float*)d_in[6];
  const float* Wpos     = (const float*)d_in[7];
  const float* bpos     = (const float*)d_in[8];
  const float* lnqg     = (const float*)d_in[9];
  const float* lnqb     = (const float*)d_in[10];
  const float* Wq1      = (const float*)d_in[11];
  const float* bq1      = (const float*)d_in[12];
  const float* Wq2      = (const float*)d_in[13];
  const float* bq2      = (const float*)d_in[14];
  const float* lnmg     = (const float*)d_in[15];
  const float* lnmb     = (const float*)d_in[16];
  const float* Wm1      = (const float*)d_in[17];
  const float* bm1      = (const float*)d_in[18];
  const float* Wm2      = (const float*)d_in[19];
  const float* bm2      = (const float*)d_in[20];
  const float* Wm3      = (const float*)d_in[21];
  const float* bm3      = (const float*)d_in[22];
  char* ws = (char*)d_ws;
  int*   pidx  = (int*)  (ws + WS_PIDX);
  float* scal  = (float*)(ws + WS_SCAL);
  float* boff  = (float*)(ws + WS_BOFF);
  float* itot  = (float*)(ws + WS_ITOT);
  float* ppart = (float*)(ws + WS_PPART);
  f16*   hws   = (f16*)  (ws + WS_H);
  f16*   wtws  = (f16*)  (ws + WS_WT);
  float* out   = (float*)d_out;

  k0_wt<<<4 * 256, 256, 0, stream>>>(Wq1, Wq2, Wm1, Wm2, wtws);
  k1_norm<<<BB * NBLK, 512, 0, stream>>>(A, mpar, nmask, out, pidx, scal, boff);
  k2_scan<<<BB, KK, 0, stream>>>(boff, itot);
  k3_pos<<<BB * NBLK * 4, 128, 0, stream>>>(A, mpar, scal, boff, itot, ppart);
  k4_emb<<<(BB * NN) / 128, 256, 0, stream>>>(s_parent, nmask, Wpos, bpos,
                                              lnqg, lnqb, pidx, ppart, hws, out);
  size_t dynBytes = (size_t)(256 * WSTR + 2 * 128 * AS) * sizeof(f16); // 270336 B
  k5_mlp<<<(BB * NN) / 128, 256, dynBytes, stream>>>(
      hws, wtws, bq1, bq2, lnmg, lnmb, bm1, bm2, Wm3, bm3,
      nmask, pidx, muk, Rk, sk, out);
}
